// HFAGNN_78271484002657
// MI455X (gfx1250) — compile-verified, matrix operand order checked by reading, not run on hardware
//
#include <hip/hip_runtime.h>

// ---------------------------------------------------------------------------
// HFAGNN forward for gfx1250 (MI455X).
//  - bf16 WMMA (v_wmma_f32_16x16x32_bf16) for all GEMMs, fp32 accumulate
//  - bf16 edge intermediates (halves dominant HBM traffic)
//  - GLOBAL_LOAD_ASYNC_TO_LDS_B128 staging for bf16 A-tiles (ASYNCcnt)
//  - fp32 global atomics for the scatter-add aggregations
// ---------------------------------------------------------------------------

#define N_NODES 40000
#define N_EDGES 640000
#define HD      128
#define MD      128
#define NL      4
#define NG      256
#define F1D     12
#define F2D     6

typedef __attribute__((ext_vector_type(16))) __bf16 v16bf;
typedef __attribute__((ext_vector_type(8)))  float  v8f;

#define FLAG_RELU  1
#define FLAG_ACCUM 2

// ------------------------- weight transpose fp32 -> bf16 -------------------
// W is [K][128] row-major (fp32);  Wt is [128][KP] bf16 (zero padded K->KP).
__global__ __launch_bounds__(256) void k_wt(const float* __restrict__ W, int K,
                                            int KP, __bf16* __restrict__ Wt) {
  int t = blockIdx.x * 256 + threadIdx.x;
  if (t >= 128 * KP) return;
  int col = t / KP, k = t % KP;
  float v = (k < K) ? W[(size_t)k * 128 + col] : 0.0f;
  Wt[t] = (__bf16)v;
}

// ------------------------- bf16 WMMA GEMM ----------------------------------
// C[R,128] = act( A[R,Kact] @ W[Kact,128] (+ bias) (+ C) )
// Block tile: 64 rows x 128 cols, 256 threads = 8 waves (wave32).
// Wave (w&3) -> row group (16 rows), (w>>2) -> col group (64 cols, 4 wmma).
// TA: float (cvt to bf16 while staging) or __bf16 (async copy to LDS).
// TC: float (supports ACCUM) or __bf16.
template <int KP, typename TA, typename TC>
__global__ __launch_bounds__(256) void k_gemm(
    const TA* __restrict__ A, int Kact, int R,
    const __bf16* __restrict__ Wt, const float* __restrict__ bias,
    TC* __restrict__ C, int flags) {
  constexpr int AP = 40;       // A-tile pitch (bf16 elems) -> 80B rows
  constexpr int WP = KP + 8;   // W pitch (bf16 elems), avoids bank conflicts
  __shared__ __bf16 sW[128 * WP];
  __shared__ __bf16 sA[64 * AP];

  const int tid   = threadIdx.x;
  const int wave  = tid >> 5;
  const int lane  = tid & 31;
  const int half  = lane >> 4;   // 0/1: which K-half of the fragment
  const int lan16 = lane & 15;
  const int row0  = blockIdx.x * 64;
  const int wm    = (wave & 3) * 16;   // wave row offset within block tile
  const int wc    = (wave >> 2) * 64;  // wave col offset

  // Stage transposed weights into LDS once (KP*128 bf16).
  for (int i = tid; i < 128 * (KP / 8); i += 256) {
    int col = i / (KP / 8);
    int kk  = (i % (KP / 8)) * 8;
    *(uint4*)&sW[col * WP + kk] = *(const uint4*)&Wt[(size_t)col * KP + kk];
  }

  v8f acc[4] = {};

  for (int k0 = 0; k0 < KP; k0 += 32) {
    __syncthreads();
    // Stage A tile: 64 rows x 32 K values (bf16), zero-padded.
    {
      int arow = tid >> 2;
      int kk   = (tid & 3) * 8;
      int gr   = row0 + arow;
      if constexpr (sizeof(TA) == 2) {
        // bf16 source: pure 16B/lane copy -> CDNA5 async global->LDS DMA.
        if (gr < R) {
          const __bf16* src =
              (const __bf16*)A + (size_t)gr * Kact + (k0 + kk);
          unsigned ldsa = (unsigned)(uintptr_t)&sA[arow * AP + kk];
          asm volatile("global_load_async_to_lds_b128 %0, %1, off"
                       :: "v"(ldsa),
                          "v"((unsigned long long)(uintptr_t)src)
                       : "memory");
        }
        asm volatile("s_wait_asynccnt 0" ::: "memory");
      } else {
        __bf16* dst = &sA[arow * AP + kk];
        alignas(16) __bf16 tmp[8];
        if (gr < R && (k0 + kk + 8) <= Kact && ((Kact & 3) == 0)) {
          const float* src = (const float*)A + (size_t)gr * Kact + (k0 + kk);
          float4 f0 = *(const float4*)(src);
          float4 f1 = *(const float4*)(src + 4);
          tmp[0] = (__bf16)f0.x; tmp[1] = (__bf16)f0.y;
          tmp[2] = (__bf16)f0.z; tmp[3] = (__bf16)f0.w;
          tmp[4] = (__bf16)f1.x; tmp[5] = (__bf16)f1.y;
          tmp[6] = (__bf16)f1.z; tmp[7] = (__bf16)f1.w;
        } else {
#pragma unroll
          for (int j = 0; j < 8; ++j) {
            int k = k0 + kk + j;
            float v =
                (gr < R && k < Kact) ? ((const float*)A)[(size_t)gr * Kact + k]
                                     : 0.0f;
            tmp[j] = (__bf16)v;
          }
        }
        *(uint4*)dst = *(const uint4*)tmp;
        if (k0 + 32 < KP && gr < R)  // prefetch next A tile slice
          __builtin_prefetch((const float*)A + (size_t)gr * Kact +
                                 (k0 + 32 + kk),
                             0, 1);
      }
    }
    __syncthreads();

    // A fragment: lane<16 holds row (wm+lan16), K = k0+{0..7,16..23};
    // lane>=16 same row, K = k0+{8..15,24..31}.
    v16bf a;
    {
      const __bf16* pa = &sA[(wm + lan16) * AP];
      *(uint4*)&a       = *(const uint4*)(pa + half * 8);
      *((uint4*)&a + 1) = *(const uint4*)(pa + 16 + half * 8);
    }
#pragma unroll
    for (int n = 0; n < 4; ++n) {
      v16bf b;
      const __bf16* pb = &sW[(wc + n * 16 + lan16) * WP + k0];
      *(uint4*)&b       = *(const uint4*)(pb + half * 8);
      *((uint4*)&b + 1) = *(const uint4*)(pb + 16 + half * 8);
      acc[n] = __builtin_amdgcn_wmma_f32_16x16x32_bf16(
          false, a, false, b, (short)0, acc[n], false, false);
    }
  }

  // Epilogue: C layout per WMMA: VGPR i -> row (i + half*8), col = lan16.
#pragma unroll
  for (int n = 0; n < 4; ++n) {
    int c = wc + n * 16 + lan16;
    float bv = bias ? bias[c] : 0.0f;
#pragma unroll
    for (int i = 0; i < 8; ++i) {
      int r = row0 + wm + i + half * 8;
      if (r < R) {
        float v = acc[n][i] + bv;
        size_t idx = (size_t)r * 128 + c;
        if constexpr (sizeof(TC) == 4) {
          if (flags & FLAG_ACCUM) v += ((const float*)C)[idx];
        }
        if (flags & FLAG_RELU) v = fmaxf(v, 0.0f);
        C[idx] = (TC)v;
      }
    }
  }
}

// ------------------------- edge scatter kernels ----------------------------
// agg[dst[e]] += h[src[e]] * ew[e]   (ew in bf16, accumulate fp32)
__global__ __launch_bounds__(256) void k_edge_mul_scatter(
    const float* __restrict__ h, const __bf16* __restrict__ ew,
    const int* __restrict__ src, const int* __restrict__ dst,
    float* __restrict__ agg, int E) {
  unsigned t = blockIdx.x * 256u + threadIdx.x;
  if (t >= (unsigned)E * 32u) return;
  int e = t >> 5;
  int c = (int)(t & 31u) * 4;
  int s = src[e], d = dst[e];
  alignas(8) __bf16 wt[4];
  *(uint2*)wt = *(const uint2*)(ew + (size_t)e * 128 + c);
  float4 hv = *(const float4*)(h + (size_t)s * 128 + c);
  float* o = agg + (size_t)d * 128 + c;
  atomicAdd(o + 0, (float)wt[0] * hv.x);
  atomicAdd(o + 1, (float)wt[1] * hv.y);
  atomicAdd(o + 2, (float)wt[2] * hv.z);
  atomicAdd(o + 3, (float)wt[3] * hv.w);
}

// agg[dst[e]] += h[src[e]] + edge_attr[e]
__global__ __launch_bounds__(256) void k_edge_add_scatter(
    const float* __restrict__ h, const float* __restrict__ ea,
    const int* __restrict__ src, const int* __restrict__ dst,
    float* __restrict__ agg, int E) {
  unsigned t = blockIdx.x * 256u + threadIdx.x;
  if (t >= (unsigned)E * 32u) return;
  int e = t >> 5;
  int c = (int)(t & 31u) * 4;
  int s = src[e], d = dst[e];
  float4 a  = *(const float4*)(ea + (size_t)e * 128 + c);
  float4 hv = *(const float4*)(h + (size_t)s * 128 + c);
  float* o = agg + (size_t)d * 128 + c;
  atomicAdd(o + 0, hv.x + a.x);
  atomicAdd(o + 1, hv.y + a.y);
  atomicAdd(o + 2, hv.z + a.z);
  atomicAdd(o + 3, hv.w + a.w);
}

// ------------------------- layernorm (wave per row) ------------------------
__global__ __launch_bounds__(256) void k_ln(const float* __restrict__ in,
                                            float* __restrict__ out,
                                            const float* __restrict__ g,
                                            const float* __restrict__ b,
                                            int R, int dorelu) {
  int wave = threadIdx.x >> 5, lane = threadIdx.x & 31;
  int r = blockIdx.x * 8 + wave;
  if (r >= R) return;
  float4 v = ((const float4*)(in + (size_t)r * 128))[lane];
  float s = v.x + v.y + v.z + v.w;
  for (int m = 16; m >= 1; m >>= 1) s += __shfl_xor(s, m, 32);
  float mu = s * (1.0f / 128.0f);
  float dx = v.x - mu, dy = v.y - mu, dz = v.z - mu, dw = v.w - mu;
  float q = dx * dx + dy * dy + dz * dz + dw * dw;
  for (int m = 16; m >= 1; m >>= 1) q += __shfl_xor(q, m, 32);
  float rstd = rsqrtf(q * (1.0f / 128.0f) + 1e-5f);
  float4 gg = ((const float4*)g)[lane];
  float4 bb = ((const float4*)b)[lane];
  float4 o;
  o.x = dx * rstd * gg.x + bb.x;
  o.y = dy * rstd * gg.y + bb.y;
  o.z = dz * rstd * gg.z + bb.z;
  o.w = dw * rstd * gg.w + bb.w;
  if (dorelu) {
    o.x = fmaxf(o.x, 0.0f); o.y = fmaxf(o.y, 0.0f);
    o.z = fmaxf(o.z, 0.0f); o.w = fmaxf(o.w, 0.0f);
  }
  ((float4*)(out + (size_t)r * 128))[lane] = o;
}

// ------------------------- elementwise helpers -----------------------------
__global__ __launch_bounds__(256) void k_gine_pre(const float* __restrict__ h,
                                                  const float* __restrict__ agg,
                                                  const float* __restrict__ eps,
                                                  int l, float* __restrict__ out,
                                                  int n4) {
  int t = blockIdx.x * 256 + threadIdx.x;
  if (t >= n4) return;
  float e = 1.0f + eps[l];
  float4 a = ((const float4*)h)[t];
  float4 b = ((const float4*)agg)[t];
  float4 o;
  o.x = e * a.x + b.x; o.y = e * a.y + b.y;
  o.z = e * a.z + b.z; o.w = e * a.w + b.w;
  ((float4*)out)[t] = o;
}

__global__ __launch_bounds__(256) void k_add(const float* __restrict__ a,
                                             const float* __restrict__ b,
                                             float* __restrict__ out, int n4) {
  int t = blockIdx.x * 256 + threadIdx.x;
  if (t >= n4) return;
  float4 x = ((const float4*)a)[t];
  float4 y = ((const float4*)b)[t];
  float4 o;
  o.x = x.x + y.x; o.y = x.y + y.y; o.z = x.z + y.z; o.w = x.w + y.w;
  ((float4*)out)[t] = o;
}

__global__ __launch_bounds__(256) void k_add_bcast(float* __restrict__ h,
                                                   const float* __restrict__ hv,
                                                   const int* __restrict__ batch,
                                                   int n) {
  unsigned t = blockIdx.x * 256u + threadIdx.x;
  if (t >= (unsigned)n * 32u) return;
  int i = t >> 5;
  int c = (int)(t & 31u) * 4;
  int g = batch[i];
  float4 a = *(const float4*)(h + (size_t)i * 128 + c);
  float4 b = *(const float4*)(hv + (size_t)g * 128 + c);
  a.x += b.x; a.y += b.y; a.z += b.z; a.w += b.w;
  *(float4*)(h + (size_t)i * 128 + c) = a;
}

__global__ __launch_bounds__(256) void k_pool(const float* __restrict__ h,
                                              const int* __restrict__ batch,
                                              float* __restrict__ pool, int n) {
  unsigned t = blockIdx.x * 256u + threadIdx.x;
  if (t >= (unsigned)n * 32u) return;
  int i = t >> 5;
  int c = (int)(t & 31u) * 4;
  int g = batch[i];
  float4 v = *(const float4*)(h + (size_t)i * 128 + c);
  float* o = pool + (size_t)g * 128 + c;
  atomicAdd(o + 0, v.x); atomicAdd(o + 1, v.y);
  atomicAdd(o + 2, v.z); atomicAdd(o + 3, v.w);
}

__global__ __launch_bounds__(256) void k_pred(const float* __restrict__ pool,
                                              const float* __restrict__ W,
                                              const float* __restrict__ b,
                                              float* __restrict__ out, int Gn) {
  int wave = threadIdx.x >> 5, lane = threadIdx.x & 31;
  int r = blockIdx.x * 8 + wave;
  if (r >= Gn) return;
  float4 v = ((const float4*)(pool + (size_t)r * 128))[lane];
  float4 w = ((const float4*)W)[lane];
  float s = v.x * w.x + v.y * w.y + v.z * w.z + v.w * w.w;
  for (int m = 16; m >= 1; m >>= 1) s += __shfl_xor(s, m, 32);
  if (lane == 0) out[r] = s + b[0];
}

// ---------------------------------------------------------------------------
extern "C" void kernel_launch(void* const* d_in, const int* in_sizes, int n_in,
                              void* d_out, int out_size, void* d_ws,
                              size_t ws_size, hipStream_t stream) {
  (void)in_sizes; (void)out_size; (void)ws_size;
  const int N = N_NODES, E = N_EDGES, H = HD, M = MD, L = NL, G = NG;

  // ---------------- input pointers ----------------
  const float *x, *edge_attr, *feat1, *feat2;
  const int *edge_index, *pos_edge_index, *batch;
  const float *f1_W1, *f1_b1, *f1_W2, *f1_b2, *f2_W1, *f2_b1, *f2_W2, *f2_b2;
  const float *c1_Wrel, *c1_brel, *c1_Wroot, *c2_Wrel, *c2_brel, *c2_Wroot;
  const float *cat_W, *cat_b, *g_eps, *g_W1, *g_b1, *g_ln_g, *g_ln_b, *g_W2, *g_b2;
  const float *ln_g, *ln_b, *v_W1, *v_b1, *v_ln_g, *v_ln_b, *v_W2, *v_b2;
  const float *pred_W, *pred_b;

  if (n_in >= 40) {
    // JAX pytree flattening: alphabetical keys, params leaves alphabetical.
    batch = (const int*)d_in[0];
    edge_attr = (const float*)d_in[1];
    edge_index = (const int*)d_in[2];
    feat1 = (const float*)d_in[3];
    feat2 = (const float*)d_in[4];
    c1_Wrel = (const float*)d_in[5]; c1_Wroot = (const float*)d_in[6];
    c1_brel = (const float*)d_in[7];
    c2_Wrel = (const float*)d_in[8]; c2_Wroot = (const float*)d_in[9];
    c2_brel = (const float*)d_in[10];
    cat_W = (const float*)d_in[11]; cat_b = (const float*)d_in[12];
    f1_W1 = (const float*)d_in[13]; f1_W2 = (const float*)d_in[14];
    f1_b1 = (const float*)d_in[15]; f1_b2 = (const float*)d_in[16];
    f2_W1 = (const float*)d_in[17]; f2_W2 = (const float*)d_in[18];
    f2_b1 = (const float*)d_in[19]; f2_b2 = (const float*)d_in[20];
    g_W1 = (const float*)d_in[21]; g_W2 = (const float*)d_in[22];
    g_b1 = (const float*)d_in[23]; g_b2 = (const float*)d_in[24];
    g_eps = (const float*)d_in[25];
    g_ln_b = (const float*)d_in[26]; g_ln_g = (const float*)d_in[27];
    ln_b = (const float*)d_in[28]; ln_g = (const float*)d_in[29];
    pred_W = (const float*)d_in[30]; pred_b = (const float*)d_in[31];
    v_W1 = (const float*)d_in[32]; v_W2 = (const float*)d_in[33];
    v_b1 = (const float*)d_in[34]; v_b2 = (const float*)d_in[35];
    v_ln_b = (const float*)d_in[36]; v_ln_g = (const float*)d_in[37];
    pos_edge_index = (const int*)d_in[38];
    x = (const float*)d_in[39];
  } else {
    // Fallback: setup_inputs() insertion order, params as one fp32 blob.
    x = (const float*)d_in[0];
    edge_attr = (const float*)d_in[1];
    feat1 = (const float*)d_in[2];
    feat2 = (const float*)d_in[3];
    const float* p = (const float*)d_in[4];
    edge_index = (const int*)d_in[5];
    pos_edge_index = (const int*)d_in[6];
    batch = (const int*)d_in[7];
    size_t o = 0;
    auto nxt = [&](size_t n) { const float* q = p + o; o += n; return q; };
    f1_W1 = nxt((size_t)L * F1D * M); f1_b1 = nxt((size_t)L * M);
    f1_W2 = nxt((size_t)L * M * H);   f1_b2 = nxt((size_t)L * H);
    f2_W1 = nxt((size_t)L * F2D * M); f2_b1 = nxt((size_t)L * M);
    f2_W2 = nxt((size_t)L * M * H);   f2_b2 = nxt((size_t)L * H);
    c1_Wrel = nxt((size_t)L * H * H); c1_brel = nxt((size_t)L * H);
    c1_Wroot = nxt((size_t)L * H * H);
    c2_Wrel = nxt((size_t)L * H * H); c2_brel = nxt((size_t)L * H);
    c2_Wroot = nxt((size_t)L * H * H);
    cat_W = nxt((size_t)L * 2 * H * H); cat_b = nxt((size_t)L * H);
    g_eps = nxt(L);
    g_W1 = nxt((size_t)L * H * H); g_b1 = nxt((size_t)L * H);
    g_ln_g = nxt((size_t)L * H);   g_ln_b = nxt((size_t)L * H);
    g_W2 = nxt((size_t)L * H * H); g_b2 = nxt((size_t)L * H);
    ln_g = nxt((size_t)L * H); ln_b = nxt((size_t)L * H);
    v_W1 = nxt((size_t)(L - 1) * H * M); v_b1 = nxt((size_t)(L - 1) * M);
    v_ln_g = nxt((size_t)(L - 1) * M);   v_ln_b = nxt((size_t)(L - 1) * M);
    v_W2 = nxt((size_t)(L - 1) * M * H); v_b2 = nxt((size_t)(L - 1) * H);
    pred_W = nxt(H); pred_b = nxt(1);
  }

  const int* esrc = edge_index;
  const int* edst = edge_index + E;
  const int* psrc = pos_edge_index;
  const int* pdst = pos_edge_index + E;

  // ---------------- workspace layout ----------------
  char* wsp = (char*)d_ws;
  size_t off = 0;
  auto take = [&](size_t bytes) -> void* {
    void* q = wsp + off;
    off += (bytes + 255) & ~(size_t)255;
    return q;
  };
  __bf16* EB1 = (__bf16*)take((size_t)E * M * 2);  // edge MLP hidden (bf16)
  __bf16* EB2 = (__bf16*)take((size_t)E * H * 2);  // edge weights (bf16)
  const size_t NB = (size_t)N * H * 4;
  float* Hin  = (float*)take(NB);
  float* Hout = (float*)take(NB);
  float* Hb   = (float*)take(NB);
  float* Agg  = (float*)take(NB);
  float* Nh1  = (float*)take(NB);
  float* Nh2  = (float*)take(NB);
  float* Nhc  = (float*)take(NB);
  float* Nt2  = (float*)take(NB);
  float* Nt3  = (float*)take(NB);
  float* PoolB = (float*)take((size_t)G * H * 4);
  float* Hvirt = (float*)take((size_t)G * H * 4);
  float* VB0   = (float*)take((size_t)G * H * 4);
  float* VT1   = (float*)take((size_t)G * M * 4);

  __bf16 *WT_f1W1[NL], *WT_f1W2[NL], *WT_f2W1[NL], *WT_f2W2[NL];
  __bf16 *WT_c1rel[NL], *WT_c1root[NL], *WT_c2rel[NL], *WT_c2root[NL];
  __bf16 *WT_cat0[NL], *WT_cat1[NL], *WT_gW1[NL], *WT_gW2[NL];
  __bf16 *WT_vW1[NL - 1], *WT_vW2[NL - 1];
  for (int l = 0; l < L; ++l) {
    WT_f1W1[l] = (__bf16*)take(128 * 32 * 2);
    WT_f1W2[l] = (__bf16*)take(128 * 128 * 2);
    WT_f2W1[l] = (__bf16*)take(128 * 32 * 2);
    WT_f2W2[l] = (__bf16*)take(128 * 128 * 2);
    WT_c1rel[l] = (__bf16*)take(128 * 128 * 2);
    WT_c1root[l] = (__bf16*)take(128 * 128 * 2);
    WT_c2rel[l] = (__bf16*)take(128 * 128 * 2);
    WT_c2root[l] = (__bf16*)take(128 * 128 * 2);
    WT_cat0[l] = (__bf16*)take(128 * 128 * 2);
    WT_cat1[l] = (__bf16*)take(128 * 128 * 2);
    WT_gW1[l] = (__bf16*)take(128 * 128 * 2);
    WT_gW2[l] = (__bf16*)take(128 * 128 * 2);
  }
  for (int l = 0; l < L - 1; ++l) {
    WT_vW1[l] = (__bf16*)take(128 * 128 * 2);
    WT_vW2[l] = (__bf16*)take(128 * 128 * 2);
  }

  // ---------------- weight prep ----------------
  auto prep = [&](const float* W, int K, int KP, __bf16* Wt) {
    int total = 128 * KP;
    k_wt<<<(total + 255) / 256, 256, 0, stream>>>(W, K, KP, Wt);
  };
  for (int l = 0; l < L; ++l) {
    prep(f1_W1 + (size_t)l * F1D * M, F1D, 32, WT_f1W1[l]);
    prep(f1_W2 + (size_t)l * M * H, 128, 128, WT_f1W2[l]);
    prep(f2_W1 + (size_t)l * F2D * M, F2D, 32, WT_f2W1[l]);
    prep(f2_W2 + (size_t)l * M * H, 128, 128, WT_f2W2[l]);
    prep(c1_Wrel + (size_t)l * H * H, 128, 128, WT_c1rel[l]);
    prep(c1_Wroot + (size_t)l * H * H, 128, 128, WT_c1root[l]);
    prep(c2_Wrel + (size_t)l * H * H, 128, 128, WT_c2rel[l]);
    prep(c2_Wroot + (size_t)l * H * H, 128, 128, WT_c2root[l]);
    prep(cat_W + (size_t)l * 2 * H * H, 128, 128, WT_cat0[l]);
    prep(cat_W + (size_t)l * 2 * H * H + (size_t)H * H, 128, 128, WT_cat1[l]);
    prep(g_W1 + (size_t)l * H * H, 128, 128, WT_gW1[l]);
    prep(g_W2 + (size_t)l * H * H, 128, 128, WT_gW2[l]);
  }
  for (int l = 0; l < L - 1; ++l) {
    prep(v_W1 + (size_t)l * H * M, 128, 128, WT_vW1[l]);
    prep(v_W2 + (size_t)l * M * H, 128, 128, WT_vW2[l]);
  }

  // ---------------- launch helpers ----------------
  const int EBLK = (int)(((long)E * 32 + 255) / 256);   // edge vec blocks
  const int NVEC4 = N * H / 4;                          // node float4 count
  const int NVBLK = (NVEC4 + 255) / 256;
  const int GVEC4 = G * H / 4;

  // node GEMM: fp32 A -> fp32 C (supports ACCUM)
  auto gemmN = [&](const float* A, int R, const __bf16* Wt, const float* bias,
                   float* C, int flags) {
    k_gemm<128, float, float>
        <<<(R + 63) / 64, 256, 0, stream>>>(A, 128, R, Wt, bias, C, flags);
  };
  // edge MLP layer 1: fp32 feat (K<=12) -> bf16 hidden
  auto gemmE1 = [&](const float* A, int Kact, const __bf16* Wt,
                    const float* bias, __bf16* C, int flags) {
    k_gemm<32, float, __bf16>
        <<<(E + 63) / 64, 256, 0, stream>>>(A, Kact, E, Wt, bias, C, flags);
  };
  // edge MLP layer 2: bf16 hidden (async LDS staging) -> bf16 edge weights
  auto gemmE2 = [&](const __bf16* A, const __bf16* Wt, const float* bias,
                    __bf16* C, int flags) {
    k_gemm<128, __bf16, __bf16>
        <<<(E + 63) / 64, 256, 0, stream>>>(A, 128, E, Wt, bias, C, flags);
  };
  auto pool_nodes = [&](const float* Hsrc) {
    hipMemsetAsync(PoolB, 0, (size_t)G * H * 4, stream);
    k_pool<<<(int)(((long)N * 32 + 255) / 256), 256, 0, stream>>>(Hsrc, batch,
                                                                  PoolB, N);
  };

  auto hybrid = [&](int l, const float* X, float* Y) {
    // --- feat1 MLP -> edge weights, EdgeGraphConv #1 ---
    gemmE1(feat1, F1D, WT_f1W1[l], f1_b1 + (size_t)l * M, EB1, FLAG_RELU);
    gemmE2(EB1, WT_f1W2[l], f1_b2 + (size_t)l * H, EB2, 0);
    hipMemsetAsync(Agg, 0, NB, stream);
    k_edge_mul_scatter<<<EBLK, 256, 0, stream>>>(X, EB2, psrc, pdst, Agg, E);
    gemmN(Agg, N, WT_c1rel[l], c1_brel + (size_t)l * H, Nh1, 0);
    gemmN(X, N, WT_c1root[l], nullptr, Nh1, FLAG_ACCUM | FLAG_RELU);
    // --- feat2 MLP, EdgeGraphConv #2 ---
    gemmE1(feat2, F2D, WT_f2W1[l], f2_b1 + (size_t)l * M, EB1, FLAG_RELU);
    gemmE2(EB1, WT_f2W2[l], f2_b2 + (size_t)l * H, EB2, 0);
    hipMemsetAsync(Agg, 0, NB, stream);
    k_edge_mul_scatter<<<EBLK, 256, 0, stream>>>(X, EB2, psrc, pdst, Agg, E);
    gemmN(Agg, N, WT_c2rel[l], c2_brel + (size_t)l * H, Nh2, 0);
    gemmN(X, N, WT_c2root[l], nullptr, Nh2, FLAG_ACCUM | FLAG_RELU);
    // --- hc = relu([h1, h2] @ cat_W + cat_b), split over K ---
    gemmN(Nh1, N, WT_cat0[l], cat_b + (size_t)l * H, Nhc, 0);
    gemmN(Nh2, N, WT_cat1[l], nullptr, Nhc, FLAG_ACCUM | FLAG_RELU);
    // --- GINEConv ---
    hipMemsetAsync(Agg, 0, NB, stream);
    k_edge_add_scatter<<<EBLK, 256, 0, stream>>>(X, edge_attr, esrc, edst, Agg, E);
    k_gine_pre<<<NVBLK, 256, 0, stream>>>(X, Agg, g_eps, l, Nt2, NVEC4);
    gemmN(Nt2, N, WT_gW1[l], g_b1 + (size_t)l * H, Nt3, 0);
    k_ln<<<(N + 7) / 8, 256, 0, stream>>>(Nt3, Nt3, g_ln_g + (size_t)l * H,
                                          g_ln_b + (size_t)l * H, N, 1);
    gemmN(Nt3, N, WT_gW2[l], g_b2 + (size_t)l * H, Nt2, 0);
    k_add<<<NVBLK, 256, 0, stream>>>(Nhc, Nt2, Y, NVEC4);
  };

  auto vmlp = [&](int vl, const float* Vin, float* target, int accumFlag) {
    gemmN(Vin, G, WT_vW1[vl], v_b1 + (size_t)vl * M, VT1, 0);
    k_ln<<<(G + 7) / 8, 256, 0, stream>>>(VT1, VT1, v_ln_g + (size_t)vl * M,
                                          v_ln_b + (size_t)vl * M, G, 1);
    gemmN(VT1, G, WT_vW2[vl], v_b2 + (size_t)vl * H, target, accumFlag);
  };

  // ---------------- forward ----------------
  hybrid(0, x, Hin);           // h_in = hybrid(0, x)
  pool_nodes(Hin);
  vmlp(0, PoolB, Hvirt, 0);    // h_virt = vmlp(0, seg_g(h_in))

  for (int layer = 1; layer < L; ++layer) {
    k_add_bcast<<<(int)(((long)N * 32 + 255) / 256), 256, 0, stream>>>(
        Hin, Hvirt, batch, N);                        // h_in += h_virt[batch]
    k_ln<<<(N + 7) / 8, 256, 0, stream>>>(Hin, Hb, ln_g + (size_t)layer * H,
                                          ln_b + (size_t)layer * H, N, 1);
    hybrid(layer, Hb, Hout);                          // h = hybrid(layer, h)
    if (layer < L - 1) {
      pool_nodes(Hout);
      k_add<<<(GVEC4 + 255) / 256, 256, 0, stream>>>(Hvirt, PoolB, VB0, GVEC4);
      vmlp(layer, VB0, Hvirt, FLAG_ACCUM);            // h_virt += vmlp(...)
    }
    k_add<<<NVBLK, 256, 0, stream>>>(Hout, Hin, Hout, NVEC4);  // h += h_in
    float* t = Hin; Hin = Hout; Hout = t;                      // h_in = h
  }

  k_ln<<<(N + 7) / 8, 256, 0, stream>>>(Hin, Hb, ln_g, ln_b, N, 1);
  pool_nodes(Hb);
  k_pred<<<(G + 7) / 8, 256, 0, stream>>>(PoolB, pred_W, pred_b,
                                          (float*)d_out, G);
}
